// QMLPerfTinyRN08_84713934946643
// MI455X (gfx1250) — 16-bit and 8-bit matrix operand order repaired and checked by reading, compiled, not run on hardware
//
#include <hip/hip_runtime.h>


typedef __attribute__((ext_vector_type(8)))  int       v8i;
typedef __attribute__((ext_vector_type(8)))  float     v8f;
typedef __attribute__((ext_vector_type(16))) _Float16  v16h;

static constexpr int BATCH = 2048;

static constexpr size_t OFF_AMAX   = 0;
static constexpr size_t OFF_SCALEA = 256;
static constexpr size_t OFF_SHIFTB = 1024;
static constexpr size_t OFF_FC_SW  = 2048;
static constexpr size_t OFF_FC_SC  = 2112;
static constexpr size_t OFF_FC_SH  = 2176;
static constexpr size_t OFF_B1     = 4096;
static constexpr size_t OFF_B2     = 8192;
static constexpr size_t OFF_B3     = 16384;
static constexpr size_t OFF_B4     = 20480;
static constexpr size_t OFF_B5     = 32768;
static constexpr size_t OFF_BFC    = 65536;
static constexpr size_t OFF_XQ     = 1ull  << 20;
static constexpr size_t OFF_H      = 18ull << 20;
static constexpr size_t OFF_QA     = 146ull << 20;
static constexpr size_t OFF_QB     = 210ull << 20;

__device__ __forceinline__ float scale_from(const unsigned int* amax, float n) {
    return fmaxf(__uint_as_float(*amax) / n, 1e-8f);
}

union V8U  { uint2 u2[4]; v8i  v; };
union V16U { uint4 u4[2]; v16h h; };

__device__ __forceinline__ v8i load_a8(const signed char* ap) {
    V8U u;
    u.u2[0] = *(const uint2*)(ap);
    u.u2[1] = *(const uint2*)(ap + 16);
    u.u2[2] = *(const uint2*)(ap + 32);
    u.u2[3] = *(const uint2*)(ap + 48);
    return u.v;
}

__global__ void init_ws_kernel(unsigned int* amax) {
    if (threadIdx.x < 8) amax[threadIdx.x] = 0u;
}

__global__ void absmax_f32_kernel(const float* __restrict__ x, size_t n,
                                  unsigned int* __restrict__ amax) {
    size_t i = (size_t)blockIdx.x * blockDim.x + threadIdx.x;
    size_t stride = (size_t)gridDim.x * blockDim.x;
    float m = 0.f;
    for (; i < n; i += stride) m = fmaxf(m, fabsf(x[i]));
    #pragma unroll
    for (int off = 16; off; off >>= 1) m = fmaxf(m, __shfl_xor(m, off, 32));
    if ((threadIdx.x & 31) == 0) atomicMax(amax, __float_as_uint(m));
}

__global__ void quant_input_kernel(const float* __restrict__ x,
                                   _Float16* __restrict__ xq,
                                   const unsigned int* __restrict__ amax) {
    float s = scale_from(amax, 32767.f);
    float inv = 1.f / s;
    size_t n = (size_t)BATCH * 32 * 32 * 4;
    size_t i = (size_t)blockIdx.x * blockDim.x + threadIdx.x;
    size_t stride = (size_t)gridDim.x * blockDim.x;
    for (; i < n; i += stride) {
        int c = (int)(i & 3);
        size_t p = i >> 2;
        int w = (int)(p & 31), h = (int)((p >> 5) & 31);
        size_t b = p >> 10;
        float v = 0.f;
        if (c < 3) {
            v = x[((b * 3 + c) * 32 + h) * 32 + w];
            float q = rintf(v * inv);
            q = fminf(fmaxf(q, -32767.f), 32767.f);
            v = q * s;
        }
        xq[i] = (_Float16)v;
    }
}

__global__ void requant_kernel(const _Float16* __restrict__ h,
                               signed char* __restrict__ q, size_t n,
                               const unsigned int* __restrict__ amax) {
    float inv = 1.f / scale_from(amax, 63.f);
    size_t i = (size_t)blockIdx.x * blockDim.x + threadIdx.x;
    size_t stride = (size_t)gridDim.x * blockDim.x;
    for (; i < n; i += stride) {
        float qi = rintf((float)h[i] * inv);
        qi = fminf(fmaxf(qi, -63.f), 63.f);
        q[i] = (signed char)(int)qi;
    }
}

__global__ void prep1_kernel(const float* __restrict__ w, const float* __restrict__ b,
                             const float* __restrict__ g, const float* __restrict__ be,
                             const float* __restrict__ m, const float* __restrict__ v,
                             unsigned int* __restrict__ amax,
                             _Float16* __restrict__ bmat,
                             float* __restrict__ scaleA, float* __restrict__ shiftB) {
    int c = threadIdx.x;
    if (c == 0) amax[1] = 0u;
    float s0 = scale_from(amax + 0, 32767.f);
    float wmax = 0.f;
    for (int k = 0; k < 27; ++k) wmax = fmaxf(wmax, fabsf(w[c * 27 + k]));
    float sw = fmaxf(wmax / 7.f, 1e-8f);
    float inv = g[c] * rsqrtf(v[c] + 1e-5f);
    float beta = be[c] - m[c] * inv;
    float bs = s0 * sw;
    float bq = rintf(b[c] / bs) * bs;
    scaleA[c] = inv;
    shiftB[c] = bq * inv + beta;
    int nt = c >> 4, n = c & 15;
    for (int k = 0; k < 64; ++k) {
        float qv = 0.f;
        if (k < 36) {
            int ci = k & 3, kk = k >> 2;
            if (ci < 3) {
                int kh = kk / 3, kw2 = kk % 3;
                float wv = w[((c * 3 + ci) * 3 + kh) * 3 + kw2];
                float qq = rintf(wv / sw);
                qq = fminf(fmaxf(qq, -7.f), 7.f);
                qv = qq * sw;
            }
        }
        int ks = k >> 5, local = k & 31;
        int laneHalf = (local >> 3) & 1, rem = (local & 7) + ((local >> 4) << 3);
        int dv = rem >> 1, par = rem & 1;
        int lane = laneHalf * 16 + n;
        bmat[(((size_t)ks * 2 + nt) * 32 + lane) * 16 + dv * 2 + par] = (_Float16)qv;
    }
}

template <int CIN, int KSTEPS, int NTILES, int COUT>
__global__ void prep_iu8_kernel(const float* __restrict__ w, const float* __restrict__ b,
                                const float* __restrict__ g, const float* __restrict__ be,
                                const float* __restrict__ m, const float* __restrict__ v,
                                const unsigned int* __restrict__ amax_prev,
                                unsigned int* __restrict__ amax_cur,
                                signed char* __restrict__ bmat,
                                float* __restrict__ scaleA, float* __restrict__ shiftB) {
    int t = threadIdx.x;
    if (t == 0) *amax_cur = 0u;
    float sa = scale_from(amax_prev, 63.f);
    constexpr int KREAL = 9 * CIN;
    float sw = 1.f;
    if (t < COUT) {
        float wmax = 0.f;
        for (int k = 0; k < KREAL; ++k) wmax = fmaxf(wmax, fabsf(w[t * KREAL + k]));
        sw = fmaxf(wmax / 7.f, 1e-8f);
        float inv = g[t] * rsqrtf(v[t] + 1e-5f);
        float beta = be[t] - m[t] * inv;
        float bs = sa * sw;
        float bq = rintf(b[t] / bs) * bs;
        scaleA[t] = sa * sw * inv;
        shiftB[t] = bq * inv + beta;
    }
    if (t >= NTILES * 16) return;
    int nt = t >> 4, n = t & 15;
    for (int k = 0; k < KSTEPS * 64; ++k) {
        int q = 0;
        if (t < COUT && k < KREAL) {
            int ci = k % CIN, kk = k / CIN;
            int kh = kk / 3, kw2 = kk % 3;
            float wv = w[((t * CIN + ci) * 3 + kh) * 3 + kw2];
            float qq = rintf(wv / sw);
            qq = fminf(fmaxf(qq, -7.f), 7.f);
            q = (int)qq;
        }
        int ks = k >> 6, local = k & 63;
        int laneHalf = (local >> 3) & 1, within = (local & 7) + ((local >> 4) << 3);
        int dv = within >> 2, byte = within & 3;
        int lane = laneHalf * 16 + n;
        bmat[(((size_t)ks * NTILES + nt) * 32 + lane) * 32 + dv * 4 + byte] = (signed char)q;
    }
}

__global__ __launch_bounds__(128) void conv1_f16_kernel(
        const _Float16* __restrict__ xin,
        const _Float16* __restrict__ bmat,
        const float* __restrict__ scaleA, const float* __restrict__ shiftB,
        _Float16* __restrict__ hout,
        unsigned int* __restrict__ amax) {
    __shared__ __align__(16) _Float16 lds[4 * 16 * 64];
    int wave = threadIdx.x >> 5, lane = threadIdx.x & 31;
    int b = blockIdx.x >> 4;
    int h = (blockIdx.x & 15) * 2 + (wave >> 1);
    int col0 = (wave & 1) * 16;
    _Float16* my = lds + wave * 16 * 64;
    const _Float16* inb = xin + (size_t)b * (32 * 32 * 4);
    for (int seg = lane; seg < 144; seg += 32) {
        int px = seg / 9, kk = seg % 9;
        int kh = kk / 3, kw2 = kk % 3;
        int ih = h + kh - 1, iw = col0 + px + kw2 - 1;
        uint2* dst = (uint2*)(my + px * 64 + kk * 4);
        if (ih >= 0 && ih < 32 && iw >= 0 && iw < 32)
            *dst = *(const uint2*)(inb + ((size_t)ih * 32 + iw) * 4);
        else
            *dst = uint2{0u, 0u};
    }
    for (int t = lane; t < 16 * 14; t += 32) {
        int px = t / 14, d = t % 14;
        *(unsigned int*)((char*)my + px * 128 + 72 + 4 * d) = 0u;
    }
    __syncthreads();

    int mcol = lane & 15, hi = lane >> 4;
    const char* abase = (const char*)my + mcol * 128;
    v8f acc[2] = {};
    #pragma unroll
    for (int ks = 0; ks < 2; ++ks) {
        const char* ap = abase + ks * 64 + hi * 16;
        V16U au;
        au.u4[0] = *(const uint4*)(ap);
        au.u4[1] = *(const uint4*)(ap + 32);
        #pragma unroll
        for (int nt = 0; nt < 2; ++nt) {
            v16h bf = *(const v16h*)(bmat + (((size_t)ks * 2 + nt) * 32 + lane) * 16);
            acc[nt] = __builtin_amdgcn_wmma_f32_16x16x32_f16(
                false, au.h, false, bf, (short)0, acc[nt], false, false);
        }
    }
    float lm = 0.f;
    #pragma unroll
    for (int nt = 0; nt < 2; ++nt) {
        int c = nt * 16 + mcol;
        float sA = scaleA[c], sB = shiftB[c];
        #pragma unroll
        for (int r = 0; r < 8; ++r) {
            int M = r + hi * 8;
            float y = acc[nt][r] * sA + sB;
            y = y > 0.f ? y : 0.f;
            hout[(((size_t)b * 32 + h) * 32 + (col0 + M)) * 32 + c] = (_Float16)y;
            lm = fmaxf(lm, y);
        }
    }
    #pragma unroll
    for (int off = 16; off; off >>= 1) lm = fmaxf(lm, __shfl_xor(lm, off, 32));
    if (lane == 0) atomicMax(amax, __float_as_uint(lm));
}

template <int CINP, int KSTEPS, int NTILES, int COUT>
__global__ __launch_bounds__(128) void conv_iu8_kernel(
        const signed char* __restrict__ qin,
        const signed char* __restrict__ bmat,
        const float* __restrict__ scaleA, const float* __restrict__ shiftB,
        _Float16* __restrict__ hout,
        unsigned int* __restrict__ amax) {
    constexpr int ROWB = KSTEPS * 64;
    __shared__ __align__(16) signed char lds[4 * 16 * ROWB];
    int wave = threadIdx.x >> 5, lane = threadIdx.x & 31;
    int b = blockIdx.x >> 4;
    int h = (blockIdx.x & 15) * 2 + (wave >> 1);
    int col0 = (wave & 1) * 16;
    signed char* my = lds + wave * 16 * ROWB;
    const signed char* inb = qin + (size_t)b * (32 * 32 * CINP);
    for (int seg = lane; seg < 144; seg += 32) {
        int px = seg / 9, kk = seg % 9;
        int kh = kk / 3, kw2 = kk % 3;
        int ih = h + kh - 1, iw = col0 + px + kw2 - 1;
        signed char* dst = my + px * ROWB + kk * CINP;
        bool inside = (ih >= 0 && ih < 32 && iw >= 0 && iw < 32);
        const signed char* src = inb + ((size_t)ih * 32 + iw) * CINP;
        if constexpr (CINP == 32) {
            int4 z{0, 0, 0, 0};
            ((int4*)dst)[0] = inside ? ((const int4*)src)[0] : z;
            ((int4*)dst)[1] = inside ? ((const int4*)src)[1] : z;
        } else {
            *(int*)dst = inside ? *(const int*)src : 0;
        }
    }
    constexpr int KREAL = 9 * CINP;
    constexpr int PADW = (ROWB - KREAL) / 4;
    for (int t = lane; t < 16 * PADW; t += 32) {
        int px = t / PADW, d = t % PADW;
        *(int*)(my + px * ROWB + KREAL + 4 * d) = 0;
    }
    __syncthreads();

    int mcol = lane & 15, hi = lane >> 4;
    const signed char* abase = my + mcol * ROWB + hi * 8;
    v8i acc[NTILES] = {};
    #pragma unroll
    for (int ks = 0; ks < KSTEPS; ++ks) {
        v8i a = load_a8(abase + ks * 64);
        #pragma unroll
        for (int nt = 0; nt < NTILES; ++nt) {
            v8i bf = *(const v8i*)(bmat + (((size_t)ks * NTILES + nt) * 32 + lane) * 32);
            acc[nt] = __builtin_amdgcn_wmma_i32_16x16x64_iu8(
                true, a, true, bf, acc[nt], false, false);
        }
    }
    float lm = 0.f;
    #pragma unroll
    for (int nt = 0; nt < NTILES; ++nt) {
        int c = nt * 16 + mcol;
        bool valid = (c < COUT);
        float sA = valid ? scaleA[c] : 0.f;
        float sB = valid ? shiftB[c] : 0.f;
        #pragma unroll
        for (int r = 0; r < 8; ++r) {
            int M = r + hi * 8;
            float y = (float)acc[nt][r] * sA + sB;
            y = y > 0.f ? y : 0.f;
            if (valid) {
                hout[(((size_t)b * 32 + h) * 32 + (col0 + M)) * COUT + c] = (_Float16)y;
                lm = fmaxf(lm, y);
            }
        }
    }
    #pragma unroll
    for (int off = 16; off; off >>= 1) lm = fmaxf(lm, __shfl_xor(lm, off, 32));
    if (lane == 0) atomicMax(amax, __float_as_uint(lm));
}

__global__ void fc_scale_kernel(const float* __restrict__ fcw, const float* __restrict__ fcb,
                                const unsigned int* __restrict__ amax5,
                                float* __restrict__ sw, float* __restrict__ sc,
                                float* __restrict__ sh) {
    __shared__ float red[256];
    int tid = threadIdx.x;
    for (int n = 0; n < 10; ++n) {
        float mx = 0.f;
        for (int k = tid; k < 32768; k += 256)
            mx = fmaxf(mx, fabsf(fcw[(size_t)n * 32768 + k]));
        red[tid] = mx;
        __syncthreads();
        for (int s = 128; s; s >>= 1) {
            if (tid < s) red[tid] = fmaxf(red[tid], red[tid + s]);
            __syncthreads();
        }
        if (tid == 0) sw[n] = fmaxf(red[0] / 7.f, 1e-8f);
        __syncthreads();
    }
    if (tid < 16) {
        float sa5 = scale_from(amax5, 63.f);
        if (tid < 10) {
            float s = sa5 * sw[tid];
            sc[tid] = s;
            sh[tid] = rintf(fcb[tid] / s) * s;
        } else {
            sw[tid] = 1.f; sc[tid] = 0.f; sh[tid] = 0.f;
        }
    }
}

__global__ void fc_prep_kernel(const float* __restrict__ fcw,
                               const float* __restrict__ sw,
                               signed char* __restrict__ bmat) {
    size_t total = 512 * 32 * 8;
    size_t i = (size_t)blockIdx.x * blockDim.x + threadIdx.x;
    size_t stride = (size_t)gridDim.x * blockDim.x;
    for (; i < total; i += stride) {
        int dv = (int)(i & 7);
        int lane = (int)((i >> 3) & 31);
        int ks = (int)(i >> 8);
        int n = lane & 15, laneHalf = lane >> 4;
        int group = dv >> 2, j = dv & 3;
        unsigned int dword = 0;
        if (n < 10) {
            float invs = 1.f / sw[n];
            for (int t = 0; t < 4; ++t) {
                int K = ks * 64 + (group * 2 + (j >> 1)) * 16 + laneHalf * 8 + (j & 1) * 4 + t;
                int c = K & 31, hw = K >> 5;
                int hh = hw >> 5, ww = hw & 31;
                int kref = c * 1024 + hh * 32 + ww;
                float qq = rintf(fcw[(size_t)n * 32768 + kref] * invs);
                qq = fminf(fmaxf(qq, -7.f), 7.f);
                dword |= ((unsigned int)((int)qq & 0xFF)) << (8 * t);
            }
        }
        *(unsigned int*)(bmat + i * 4) = dword;
    }
}

__global__ __launch_bounds__(256) void fc_kernel(
        const signed char* __restrict__ qa,
        const signed char* __restrict__ bmat,
        const float* __restrict__ sc, const float* __restrict__ sh,
        float* __restrict__ out) {
    __shared__ float logits[8][16][17];
    int wave = threadIdx.x >> 5, lane = threadIdx.x & 31;
    int rowbase = (blockIdx.x * 8 + wave) * 16;
    int mcol = lane & 15, hi = lane >> 4;
    const signed char* arow = qa + (size_t)(rowbase + mcol) * 32768 + hi * 8;
    v8i acc = {};
    for (int ks = 0; ks < 512; ++ks) {
        const signed char* ap = arow + ks * 64;
        __builtin_prefetch(ap + 256, 0, 1);
        v8i a = load_a8(ap);
        v8i bf = *(const v8i*)(bmat + ((size_t)ks * 32 + lane) * 32);
        acc = __builtin_amdgcn_wmma_i32_16x16x64_iu8(true, a, true, bf, acc, false, false);
    }
    float sA = sc[mcol], sB = sh[mcol];
    #pragma unroll
    for (int r = 0; r < 8; ++r)
        logits[wave][r + hi * 8][mcol] = (float)acc[r] * sA + sB;
    __syncthreads();
    if (lane < 16) {
        int row = rowbase + lane;
        float mx = -3.4e38f;
        float l[10];
        #pragma unroll
        for (int j = 0; j < 10; ++j) { l[j] = logits[wave][lane][j]; mx = fmaxf(mx, l[j]); }
        float sum = 0.f;
        #pragma unroll
        for (int j = 0; j < 10; ++j) { l[j] = __expf(l[j] - mx); sum += l[j]; }
        float invs = 1.f / sum;
        #pragma unroll
        for (int j = 0; j < 10; ++j) out[(size_t)row * 10 + j] = l[j] * invs;
    }
}

extern "C" void kernel_launch(void* const* d_in, const int* in_sizes, int n_in,
                              void* d_out, int out_size, void* d_ws, size_t ws_size,
                              hipStream_t stream) {
    const float* x   = (const float*)d_in[0];
    const float* W[6], *Bv[6], *G[6], *BE[6], *M[6], *V[6];
    for (int i = 1; i <= 5; ++i) {
        int base = 1 + (i - 1) * 6;
        W[i]  = (const float*)d_in[base + 0];
        Bv[i] = (const float*)d_in[base + 1];
        G[i]  = (const float*)d_in[base + 2];
        BE[i] = (const float*)d_in[base + 3];
        M[i]  = (const float*)d_in[base + 4];
        V[i]  = (const float*)d_in[base + 5];
    }
    const float* fcw = (const float*)d_in[31];
    const float* fcb = (const float*)d_in[32];

    char* ws = (char*)d_ws;
    unsigned int* amax = (unsigned int*)(ws + OFF_AMAX);
    float* scaleA = (float*)(ws + OFF_SCALEA);
    float* shiftB = (float*)(ws + OFF_SHIFTB);
    float* fcsw = (float*)(ws + OFF_FC_SW);
    float* fcsc = (float*)(ws + OFF_FC_SC);
    float* fcsh = (float*)(ws + OFF_FC_SH);
    _Float16*    b1m = (_Float16*)(ws + OFF_B1);
    signed char* b2m = (signed char*)(ws + OFF_B2);
    signed char* b3m = (signed char*)(ws + OFF_B3);
    signed char* b4m = (signed char*)(ws + OFF_B4);
    signed char* b5m = (signed char*)(ws + OFF_B5);
    signed char* bfc = (signed char*)(ws + OFF_BFC);
    _Float16*    xq  = (_Float16*)(ws + OFF_XQ);
    _Float16*    hbuf = (_Float16*)(ws + OFF_H);
    signed char* qA  = (signed char*)(ws + OFF_QA);
    signed char* qB  = (signed char*)(ws + OFF_QB);

    const size_t NX   = (size_t)BATCH * 3 * 32 * 32;
    const size_t N32  = (size_t)BATCH * 32 * 32 * 32;
    const size_t N4   = (size_t)BATCH * 32 * 32 * 4;
    const int CONV_GRID = BATCH * 16;

    init_ws_kernel<<<1, 32, 0, stream>>>(amax);
    absmax_f32_kernel<<<1024, 256, 0, stream>>>(x, NX, amax + 0);
    prep1_kernel<<<1, 32, 0, stream>>>(W[1], Bv[1], G[1], BE[1], M[1], V[1],
                                       amax, b1m, scaleA + 0, shiftB + 0);
    quant_input_kernel<<<2048, 256, 0, stream>>>(x, xq, amax + 0);
    conv1_f16_kernel<<<CONV_GRID, 128, 0, stream>>>(xq, b1m, scaleA + 0, shiftB + 0,
                                                    hbuf, amax + 1);
    prep_iu8_kernel<32, 5, 1, 4><<<1, 32, 0, stream>>>(
        W[2], Bv[2], G[2], BE[2], M[2], V[2], amax + 1, amax + 2,
        b2m, scaleA + 32, shiftB + 32);
    requant_kernel<<<4096, 256, 0, stream>>>(hbuf, qA, N32, amax + 1);
    conv_iu8_kernel<32, 5, 1, 4><<<CONV_GRID, 128, 0, stream>>>(
        qA, b2m, scaleA + 32, shiftB + 32, hbuf, amax + 2);
    prep_iu8_kernel<4, 1, 2, 32><<<1, 32, 0, stream>>>(
        W[3], Bv[3], G[3], BE[3], M[3], V[3], amax + 2, amax + 3,
        b3m, scaleA + 64, shiftB + 64);
    requant_kernel<<<512, 256, 0, stream>>>(hbuf, qB, N4, amax + 2);
    conv_iu8_kernel<4, 1, 2, 32><<<CONV_GRID, 128, 0, stream>>>(
        qB, b3m, scaleA + 64, shiftB + 64, hbuf, amax + 3);
    prep_iu8_kernel<32, 5, 2, 32><<<1, 32, 0, stream>>>(
        W[4], Bv[4], G[4], BE[4], M[4], V[4], amax + 3, amax + 4,
        b4m, scaleA + 96, shiftB + 96);
    requant_kernel<<<4096, 256, 0, stream>>>(hbuf, qA, N32, amax + 3);
    conv_iu8_kernel<32, 5, 2, 32><<<CONV_GRID, 128, 0, stream>>>(
        qA, b4m, scaleA + 96, shiftB + 96, hbuf, amax + 4);
    prep_iu8_kernel<32, 5, 2, 32><<<1, 32, 0, stream>>>(
        W[5], Bv[5], G[5], BE[5], M[5], V[5], amax + 4, amax + 5,
        b5m, scaleA + 128, shiftB + 128);
    requant_kernel<<<4096, 256, 0, stream>>>(hbuf, qB, N32, amax + 4);
    conv_iu8_kernel<32, 5, 2, 32><<<CONV_GRID, 128, 0, stream>>>(
        qB, b5m, scaleA + 128, shiftB + 128, hbuf, amax + 5);
    fc_scale_kernel<<<1, 256, 0, stream>>>(fcw, fcb, amax + 5, fcsw, fcsc, fcsh);
    requant_kernel<<<4096, 256, 0, stream>>>(hbuf, qA, N32, amax + 5);
    fc_prep_kernel<<<512, 256, 0, stream>>>(fcw, fcsw, bfc);
    fc_kernel<<<16, 256, 0, stream>>>(qA, bfc, fcsc, fcsh, (float*)d_out);
    (void)in_sizes; (void)n_in; (void)out_size; (void)ws_size;
}
